// SelfAttention_231928234293
// MI455X (gfx1250) — matrix operand order repaired and checked by reading, not compiled, not run on hardware
//
#include <hip/hip_runtime.h>
#include <hip/hip_fp16.h>

// ---------------------------------------------------------------------------
// Self-attention for B=8, S=2048, D_IN=D_H=512 (fp32 in/out) on gfx1250.
// GEMMs in f16 WMMA (v_wmma_f32_16x16x32_f16), fp32 accumulation,
// flash-attention softmax (no 128MB score tensor), and CDNA5 async
// global->LDS tile streaming (GLOBAL_LOAD_ASYNC_TO_LDS_B128 + ASYNCcnt).
// ---------------------------------------------------------------------------

typedef _Float16 h8   __attribute__((ext_vector_type(8)));
typedef _Float16 v16h __attribute__((ext_vector_type(16)));
typedef float    v8f  __attribute__((ext_vector_type(8)));

#define WMMA16(a, b, c) \
    __builtin_amdgcn_wmma_f32_16x16x32_f16(false, (a), false, (b), (short)0, (c), false, false)

// CDNA5 async copy: LDS[lds_off .. +15] = MEM[gaddr .. +15], tracked by ASYNCcnt.
__device__ __forceinline__ void async_load_b128(const void* gptr, void* lptr) {
    unsigned int       lds_off = (unsigned int)(unsigned long long)lptr;  // LDS aperture low 32 bits
    unsigned long long gaddr   = (unsigned long long)gptr;
    asm volatile("global_load_async_to_lds_b128 %0, %1, off"
                 :: "v"(lds_off), "v"(gaddr) : "memory");
}
__device__ __forceinline__ void wait_asynccnt0() {
    asm volatile("s_wait_asynccnt 0x0" ::: "memory");
}

static constexpr int BATCH = 8;
static constexpr int SEQ   = 2048;
static constexpr int DIM   = 512;
static constexpr int MROWS = BATCH * SEQ;   // 16384

// ---------------------------------------------------------------------------
// Kernel 1: y = x @ W^T + bias  (M=16384, N=512, K=512), output f16, optional
// scale (used to fold 1/sqrt(D_H) into Q). 128x128 tile / block, 8 waves.
// fp32->f16 conversion happens in-register, so tiles stage through VGPRs.
// ---------------------------------------------------------------------------
__global__ __launch_bounds__(256)
void qkv_proj_kernel(const float* __restrict__ x,
                     const float* __restrict__ W,
                     const float* __restrict__ bias,
                     _Float16*    __restrict__ outh,
                     float scale)
{
    // padded LDS tiles: stride 40 halves = 80B -> 16 rows hit distinct banks
    __shared__ _Float16 xs [128 * 40];
    __shared__ _Float16 wsh[128 * 40];

    const int tid  = threadIdx.x;
    const int lane = tid & 31;
    const int w    = tid >> 5;
    const int lh   = lane >> 4;      // lane-half (0/1)
    const int ln   = lane & 15;
    const int wm   = w & 3;          // wave pos along M (4)
    const int wn   = w >> 2;         // wave pos along N (2)
    const int mrow0 = blockIdx.x * 128;
    const int ncol0 = blockIdx.y * 128;

    const v8f vzero = {};
    v8f acc[2][4];
    #pragma unroll
    for (int mi = 0; mi < 2; ++mi)
        #pragma unroll
        for (int ni = 0; ni < 4; ++ni) acc[mi][ni] = vzero;

    for (int k0 = 0; k0 < DIM; k0 += 32) {
        // ---- stage fp32 -> f16 tiles (128 rows x 32 cols each) ----
        #pragma unroll
        for (int i = 0; i < 4; ++i) {
            int lin = tid + i * 256;
            int row = lin >> 3;
            int c4  = lin & 7;
            float4 xv = ((const float4*)x)[(size_t)(mrow0 + row) * 128 + (k0 >> 2) + c4];
            float4 wv = ((const float4*)W)[(size_t)(ncol0 + row) * 128 + (k0 >> 2) + c4];
            _Float16* xd = &xs [row * 40 + c4 * 4];
            xd[0] = (_Float16)xv.x; xd[1] = (_Float16)xv.y;
            xd[2] = (_Float16)xv.z; xd[3] = (_Float16)xv.w;
            _Float16* wd = &wsh[row * 40 + c4 * 4];
            wd[0] = (_Float16)wv.x; wd[1] = (_Float16)wv.y;
            wd[2] = (_Float16)wv.z; wd[3] = (_Float16)wv.w;
        }
        __syncthreads();

        // ---- fragments + WMMA ----
        v16h afrag[2], bfrag[4];
        #pragma unroll
        for (int mi = 0; mi < 2; ++mi) {
            int m = (wm * 2 + mi) * 16 + ln;
            h8 lo = *(const h8*)&xs[m * 40 + (lh ? 8 : 0)];
            h8 hi = *(const h8*)&xs[m * 40 + 16 + (lh ? 8 : 0)];
            v16h a;
            #pragma unroll
            for (int e = 0; e < 8; ++e) { a[e] = lo[e]; a[e + 8] = hi[e]; }
            afrag[mi] = a;
        }
        #pragma unroll
        for (int ni = 0; ni < 4; ++ni) {
            int n = (wn * 4 + ni) * 16 + ln;
            h8 lo = *(const h8*)&wsh[n * 40 + (lh ? 8 : 0)];
            h8 hi = *(const h8*)&wsh[n * 40 + 16 + (lh ? 8 : 0)];
            v16h b;
            #pragma unroll
            for (int e = 0; e < 8; ++e) { b[e] = lo[e]; b[e + 8] = hi[e]; }
            bfrag[ni] = b;
        }
        #pragma unroll
        for (int mi = 0; mi < 2; ++mi)
            #pragma unroll
            for (int ni = 0; ni < 4; ++ni)
                acc[mi][ni] = WMMA16(afrag[mi], bfrag[ni], acc[mi][ni]);
        __syncthreads();
    }

    // ---- bias + scale epilogue, f16 store ----
    #pragma unroll
    for (int ni = 0; ni < 4; ++ni) {
        int colg = ncol0 + (wn * 4 + ni) * 16 + ln;
        float bv = bias[colg];
        #pragma unroll
        for (int mi = 0; mi < 2; ++mi) {
            int rowbase = mrow0 + (wm * 2 + mi) * 16 + 8 * lh;
            #pragma unroll
            for (int r = 0; r < 8; ++r)
                outh[(size_t)(rowbase + r) * DIM + colg] =
                    (_Float16)((acc[mi][ni][r] + bv) * scale);
        }
    }
}

// ---------------------------------------------------------------------------
// Kernel 2: flash attention. Block = (64 q-rows, one batch), 8 waves.
// LDS (dynamic, ~227KB): Q tile resident, K/V tiles streamed with
// GLOBAL_LOAD_ASYNC_TO_LDS_B128 (Bc = 64 keys per iteration).
// ---------------------------------------------------------------------------
static constexpr int QS_STRIDE = 520;  // halves (1040B, 16B aligned, bank-shifted)
static constexpr int SC_STRIDE = 68;   // floats
static constexpr int PP_STRIDE = 72;   // halves

static constexpr int OFF_KS = 64 * QS_STRIDE * 2;          // 66560
static constexpr int OFF_VS = OFF_KS * 2;                  // 133120
static constexpr int OFF_SC = OFF_KS * 3;                  // 199680
static constexpr int OFF_PP = OFF_SC + 64 * SC_STRIDE * 4; // 217088
static constexpr int OFF_AL = OFF_PP + 64 * PP_STRIDE * 2; // 226304
static constexpr int OFF_LL = OFF_AL + 64 * 4;             // 226560
static constexpr int LDS_BYTES = OFF_LL + 64 * 4;          // 226816

__global__ __launch_bounds__(256)
void attn_kernel(const _Float16* __restrict__ Qh,
                 const _Float16* __restrict__ Kh,
                 const _Float16* __restrict__ Vh,
                 float* __restrict__ out)
{
    extern __shared__ char smem[];
    _Float16* Qs = (_Float16*)(smem);
    _Float16* Ks = (_Float16*)(smem + OFF_KS);
    _Float16* Vs = (_Float16*)(smem + OFF_VS);
    float*    Sc = (float*)   (smem + OFF_SC);
    _Float16* Pp = (_Float16*)(smem + OFF_PP);
    float*    Al = (float*)   (smem + OFF_AL);
    float*    Ll = (float*)   (smem + OFF_LL);

    const int tid  = threadIdx.x;
    const int lane = tid & 31;
    const int w    = tid >> 5;
    const int lh   = lane >> 4;
    const int ln   = lane & 15;
    const int q0   = blockIdx.x * 64;
    const size_t base = (size_t)blockIdx.y * SEQ * DIM;

    // ---- async-load Q tile (64 x 512 f16) once; waited on in iteration 0 ----
    #pragma unroll
    for (int i = 0; i < 16; ++i) {
        int lin = tid + i * 256;
        int row = lin >> 6;
        int c   = (lin & 63) * 8;
        async_load_b128(&Qh[base + (size_t)(q0 + row) * DIM + c], &Qs[row * QS_STRIDE + c]);
    }

    float m_i = -1e30f, l_i = 0.0f;  // online softmax state (valid for tid<64)
    const v8f vzero = {};
    v8f o[4][4];
    #pragma unroll
    for (int mt = 0; mt < 4; ++mt)
        #pragma unroll
        for (int nt = 0; nt < 4; ++nt) o[mt][nt] = vzero;

    for (int j = 0; j < SEQ; j += 64) {
        __syncthreads();  // prior phase B done before K/V overwrite
        #pragma unroll
        for (int i = 0; i < 16; ++i) {
            int lin = tid + i * 256;
            int row = lin >> 6;
            int c   = (lin & 63) * 8;
            async_load_b128(&Kh[base + (size_t)(j + row) * DIM + c], &Ks[row * QS_STRIDE + c]);
            async_load_b128(&Vh[base + (size_t)(j + row) * DIM + c], &Vs[row * QS_STRIDE + c]);
        }
        wait_asynccnt0();   // this wave's tile chunks (and Q on iter 0) landed in LDS
        __syncthreads();    // ... and everyone else's

        // ---- phase A: scores S = Q*K^T  (wave: 16 rows x 32 keys) ----
        {
            const int mt = w & 3;   // q-row tile
            const int nh = w >> 2;  // key-col half
            v8f s0 = vzero, s1 = vzero;
            for (int d = 0; d < DIM; d += 32) {
                v16h a;
                {
                    int m = mt * 16 + ln;
                    h8 lo = *(const h8*)&Qs[m * QS_STRIDE + d + (lh ? 8 : 0)];
                    h8 hi = *(const h8*)&Qs[m * QS_STRIDE + d + 16 + (lh ? 8 : 0)];
                    #pragma unroll
                    for (int e = 0; e < 8; ++e) { a[e] = lo[e]; a[e + 8] = hi[e]; }
                }
                #pragma unroll
                for (int ni = 0; ni < 2; ++ni) {
                    int n = nh * 32 + ni * 16 + ln;
                    h8 lo = *(const h8*)&Ks[n * QS_STRIDE + d + (lh ? 8 : 0)];
                    h8 hi = *(const h8*)&Ks[n * QS_STRIDE + d + 16 + (lh ? 8 : 0)];
                    v16h b;
                    #pragma unroll
                    for (int e = 0; e < 8; ++e) { b[e] = lo[e]; b[e + 8] = hi[e]; }
                    if (ni == 0) s0 = WMMA16(a, b, s0);
                    else         s1 = WMMA16(a, b, s1);
                }
            }
            #pragma unroll
            for (int r = 0; r < 8; ++r) {
                int row = mt * 16 + r + 8 * lh;
                Sc[row * SC_STRIDE + nh * 32 + ln]      = s0[r];
                Sc[row * SC_STRIDE + nh * 32 + 16 + ln] = s1[r];
            }
        }
        __syncthreads();

        // ---- online softmax: thread t owns q-row t (t < 64) ----
        if (tid < 64) {
            float mj = -1e30f;
            #pragma unroll
            for (int c = 0; c < 64; ++c) mj = fmaxf(mj, Sc[tid * SC_STRIDE + c]);
            float mnew  = fmaxf(m_i, mj);
            float alpha = __expf(m_i - mnew);
            float ps = 0.0f;
            #pragma unroll
            for (int c = 0; c < 64; ++c) {
                float p = __expf(Sc[tid * SC_STRIDE + c] - mnew);
                Pp[tid * PP_STRIDE + c] = (_Float16)p;
                ps += p;
            }
            l_i = l_i * alpha + ps;
            m_i = mnew;
            Al[tid] = alpha;
        }
        __syncthreads();

        // ---- phase B: O = O*alpha + P*V  (wave: 64 rows x 64 cols) ----
        #pragma unroll
        for (int mt = 0; mt < 4; ++mt)
            #pragma unroll
            for (int r = 0; r < 8; ++r) {
                float a = Al[mt * 16 + r + 8 * lh];
                #pragma unroll
                for (int nt = 0; nt < 4; ++nt) o[mt][nt][r] *= a;
            }
        #pragma unroll
        for (int ks = 0; ks < 2; ++ks) {
            const int kb = ks * 32;
            v16h pa[4];
            #pragma unroll
            for (int mt = 0; mt < 4; ++mt) {
                int m = mt * 16 + ln;
                h8 lo = *(const h8*)&Pp[m * PP_STRIDE + kb + (lh ? 8 : 0)];
                h8 hi = *(const h8*)&Pp[m * PP_STRIDE + kb + 16 + (lh ? 8 : 0)];
                v16h a;
                #pragma unroll
                for (int e = 0; e < 8; ++e) { a[e] = lo[e]; a[e + 8] = hi[e]; }
                pa[mt] = a;
            }
            #pragma unroll
            for (int nt = 0; nt < 4; ++nt) {
                int col = w * 64 + nt * 16 + ln;
                v16h vb;
                #pragma unroll
                for (int e = 0; e < 16; ++e)
                    vb[e] = Vs[(kb + (e < 8 ? 8 * lh + e : 16 + 8 * lh + (e - 8))) * QS_STRIDE + col];
                #pragma unroll
                for (int mt = 0; mt < 4; ++mt)
                    o[mt][nt] = WMMA16(pa[mt], vb, o[mt][nt]);
            }
        }
    }

    __syncthreads();
    if (tid < 64) Ll[tid] = l_i;
    __syncthreads();

    // ---- epilogue: divide by softmax denom, fp32 store ----
    #pragma unroll
    for (int mt = 0; mt < 4; ++mt)
        #pragma unroll
        for (int r = 0; r < 8; ++r) {
            int row = mt * 16 + r + 8 * lh;
            float inv = 1.0f / Ll[row];
            #pragma unroll
            for (int nt = 0; nt < 4; ++nt) {
                int col = w * 64 + nt * 16 + ln;
                out[base + (size_t)(q0 + row) * DIM + col] = o[mt][nt][r] * inv;
            }
        }
}

// ---------------------------------------------------------------------------
extern "C" void kernel_launch(void* const* d_in, const int* in_sizes, int n_in,
                              void* d_out, int out_size, void* d_ws, size_t ws_size,
                              hipStream_t stream)
{
    (void)in_sizes; (void)n_in; (void)out_size; (void)ws_size;
    const float* x  = (const float*)d_in[0];
    const float* Wq = (const float*)d_in[1];
    const float* bq = (const float*)d_in[2];
    const float* Wk = (const float*)d_in[3];
    const float* bk = (const float*)d_in[4];
    const float* Wv = (const float*)d_in[5];
    const float* bv = (const float*)d_in[6];
    float* out = (float*)d_out;

    _Float16* Qh = (_Float16*)d_ws;
    _Float16* Kh = Qh + (size_t)MROWS * DIM;
    _Float16* Vh = Kh + (size_t)MROWS * DIM;

    dim3 g1(MROWS / 128, DIM / 128), b1(256);
    const float qscale = 0.044194173824159216f;  // 1/sqrt(512)
    qkv_proj_kernel<<<g1, b1, 0, stream>>>(x, Wq, bq, Qh, qscale);
    qkv_proj_kernel<<<g1, b1, 0, stream>>>(x, Wk, bk, Kh, 1.0f);
    qkv_proj_kernel<<<g1, b1, 0, stream>>>(x, Wv, bv, Vh, 1.0f);

    (void)hipFuncSetAttribute((const void*)attn_kernel,
                              hipFuncAttributeMaxDynamicSharedMemorySize, LDS_BYTES);
    dim3 g2(SEQ / 64, BATCH), b2(256);
    attn_kernel<<<g2, b2, LDS_BYTES, stream>>>(Qh, Kh, Vh, out);
}
